// transformAttention_45672682225726
// MI455X (gfx1250) — matrix operand order repaired and checked by reading, not compiled, not run on hardware
//
#include <hip/hip_runtime.h>

typedef __attribute__((ext_vector_type(16))) _Float16 v16h;
typedef __attribute__((ext_vector_type(8)))  float    v8f;

#define Bdim 16
#define Tdim 12        // P == Q == 12
#define Ndim 2048
#define Ddim 64
#define Hh   8         // heads
#define Hk   8         // head dim
#define NT   4         // n-values per block
#define ROWS (NT*Tdim) // 48 rows == exactly 3 M-tiles of 16
#define MT   (ROWS/16) // 3
#define STR  72        // f16 LDS row stride (padded vs 64 to dodge bank conflicts)

// ---- fragment gathers per CDNA5 WMMA VGPR layouts (CDNA5 ISA §7.12.2) ----

// A (16x32 f16): lane&15 = M; lanes>=16 hold K+8 block; VGPR 0-3: K=0..7(+8), VGPR 4-7: K=16..23(+8)
__device__ __forceinline__ v16h frag_a_lds(const _Float16* base, int m0, int lane, int k0) {
  const int m = m0 + (lane & 15);
  const int g = (lane >> 4) * 8;
  const _Float16* row = base + m * STR + k0;
  v16h a;
#pragma unroll
  for (int j = 0; j < 8; ++j) {
    const int kb = (j < 4) ? (g + 2*j) : (16 + g + 2*(j-4));
    a[2*j]   = row[kb];
    a[2*j+1] = row[kb+1];
  }
  return a;
}

// Same A layout, but from a per-lane global f32 row pointer (convert on the fly)
__device__ __forceinline__ v16h frag_a_gf32(const float* row, int lane, int k0) {
  const int g = (lane >> 4) * 8;
  v16h a;
#pragma unroll
  for (int j = 0; j < 8; ++j) {
    const int kb = k0 + ((j < 4) ? (g + 2*j) : (16 + g + 2*(j-4)));
    a[2*j]   = (_Float16)row[kb];
    a[2*j+1] = (_Float16)row[kb+1];
  }
  return a;
}

// B (32x16 f16): lane&15 = N; same K layout as A: lanes>=16 hold K+8 block; VGPR 0-3: K=0..7(+8), VGPR 4-7: K=16..23(+8).
// Weights stored TRANSPOSED in LDS (wt[n][k]) so each lane reads contiguous K pairs.
__device__ __forceinline__ v16h frag_b_wt(const _Float16* wt, int n0, int lane, int k0) {
  const _Float16* row = wt + (n0 + (lane & 15)) * STR + k0 + (lane >> 4) * 8;
  v16h b;
#pragma unroll
  for (int j = 0; j < 8; ++j) {
    const int kb = (j < 4) ? (2*j) : (2*j + 8);
    b[2*j]   = row[kb];
    b[2*j+1] = row[kb+1];
  }
  return b;
}

__global__ __launch_bounds__(256) void fused_attn_kernel(
    const float* __restrict__ x,   const float* __restrict__ steP,
    const float* __restrict__ steQ,
    const float* __restrict__ Wff, const float* __restrict__ bff,
    const float* __restrict__ W1,  const float* __restrict__ b1,
    const float* __restrict__ W2,  const float* __restrict__ b2,
    float* __restrict__ out)
{
  __shared__ _Float16 sWff[Ddim*STR], sW1[Ddim*STR], sW2[Ddim*STR]; // transposed f16 weights
  __shared__ float    sBff[Ddim], sB1[Ddim], sB2[Ddim];
  __shared__ _Float16 sQ[ROWS*STR], sK[ROWS*STR], sV[ROWS*STR];     // projected q/k/v
  __shared__ _Float16 sO[ROWS*STR], sH[ROWS*STR];                   // attn out, hidden

  const int tid  = threadIdx.x;
  const int lane = tid & 31;
  const int wave = tid >> 5;
  const int bb   = blockIdx.y;
  const int n0   = blockIdx.x * NT;

  // ---- Stage weights (transposed) + biases into LDS ----
  for (int i = tid; i < Ddim*Ddim; i += 256) {
    const int k = i >> 6, n = i & 63;
    sWff[n*STR + k] = (_Float16)Wff[i];
    sW1 [n*STR + k] = (_Float16)W1[i];
    sW2 [n*STR + k] = (_Float16)W2[i];
  }
  if (tid < Ddim) { sBff[tid] = bff[tid]; sB1[tid] = b1[tid]; sB2[tid] = b2[tid]; }
  __syncthreads();

  // ---- Phase 1: q/k/v = {STE_Q, STE_P, x} @ W_ff + b_ff  (WMMA, A from global) ----
  for (int s = wave; s < 3*MT; s += 8) {
    const int mat = s / MT, mi = s % MT;
    const float* src = (mat == 0) ? steQ : ((mat == 1) ? steP : x);
    _Float16*    dst = (mat == 0) ? sQ   : ((mat == 1) ? sK   : sV);
    const int r  = mi*16 + (lane & 15);            // block-local row this lane feeds
    const int t  = r % Tdim;
    const int nn = n0 + r / Tdim;
    const float* rowp = src + (((size_t)bb*Tdim + t)*Ndim + nn)*Ddim;
    const v16h a0 = frag_a_gf32(rowp, lane, 0);
    const v16h a1 = frag_a_gf32(rowp, lane, 32);
#pragma unroll
    for (int nt = 0; nt < 4; ++nt) {
      v8f acc = {};
      const v16h bq0 = frag_b_wt(sWff, nt*16, lane, 0);
      acc = __builtin_amdgcn_wmma_f32_16x16x32_f16(false, a0, false, bq0, (short)0, acc, false, false);
      const v16h bq1 = frag_b_wt(sWff, nt*16, lane, 32);
      acc = __builtin_amdgcn_wmma_f32_16x16x32_f16(false, a1, false, bq1, (short)0, acc, false, false);
      const int nc = nt*16 + (lane & 15);
      const int mb = (lane >> 4) * 8;
#pragma unroll
      for (int j = 0; j < 8; ++j)
        dst[(mi*16 + mb + j)*STR + nc] = (_Float16)(acc[j] + sBff[nc]);
    }
  }
  __syncthreads();

  // ---- Phase 2: per-(n, head) 12x12 attention + softmax (VALU from LDS) ----
  for (int w = tid; w < NT*Hh*Tdim; w += 256) {
    const int nl = w / (Hh*Tdim);
    const int h  = (w / Tdim) % Hh;
    const int qi = w % Tdim;
    const _Float16* qr = &sQ[(nl*Tdim + qi)*STR + h*Hk];
    float qv[Hk];
#pragma unroll
    for (int d = 0; d < Hk; ++d) qv[d] = (float)qr[d];
    float sc[Tdim], mx = -1e30f;
#pragma unroll
    for (int p = 0; p < Tdim; ++p) {
      const _Float16* kr = &sK[(nl*Tdim + p)*STR + h*Hk];
      float acc = 0.f;
#pragma unroll
      for (int d = 0; d < Hk; ++d) acc += qv[d] * (float)kr[d];
      acc *= 0.35355339059327373f;   // 1/sqrt(h), h=8
      sc[p] = acc;
      mx = fmaxf(mx, acc);
    }
    float sum = 0.f;
#pragma unroll
    for (int p = 0; p < Tdim; ++p) { sc[p] = __expf(sc[p] - mx); sum += sc[p]; }
    const float inv = 1.0f / sum;
    float ov[Hk];
#pragma unroll
    for (int d = 0; d < Hk; ++d) ov[d] = 0.f;
#pragma unroll
    for (int p = 0; p < Tdim; ++p) {
      const float a = sc[p] * inv;
      const _Float16* vr = &sV[(nl*Tdim + p)*STR + h*Hk];
#pragma unroll
      for (int d = 0; d < Hk; ++d) ov[d] += a * (float)vr[d];
    }
    _Float16* orow = &sO[(nl*Tdim + qi)*STR + h*Hk];
#pragma unroll
    for (int d = 0; d < Hk; ++d) orow[d] = (_Float16)ov[d];
  }
  __syncthreads();

  // ---- Phase 3: hid = relu(out @ W1 + b1)  (WMMA) ----
  for (int job = wave; job < MT*4; job += 8) {
    const int mi = job >> 2, nt = job & 3;
    const v16h a0 = frag_a_lds(sO, mi*16, lane, 0);
    const v16h a1 = frag_a_lds(sO, mi*16, lane, 32);
    v8f acc = {};
    const v16h w0 = frag_b_wt(sW1, nt*16, lane, 0);
    acc = __builtin_amdgcn_wmma_f32_16x16x32_f16(false, a0, false, w0, (short)0, acc, false, false);
    const v16h w1 = frag_b_wt(sW1, nt*16, lane, 32);
    acc = __builtin_amdgcn_wmma_f32_16x16x32_f16(false, a1, false, w1, (short)0, acc, false, false);
    const int nc = nt*16 + (lane & 15);
    const int mb = (lane >> 4) * 8;
#pragma unroll
    for (int j = 0; j < 8; ++j)
      sH[(mi*16 + mb + j)*STR + nc] = (_Float16)fmaxf(acc[j] + sB1[nc], 0.0f);
  }
  __syncthreads();

  // ---- Phase 4: final = hid @ W2 + b2  (WMMA, f32 store to global) ----
  for (int job = wave; job < MT*4; job += 8) {
    const int mi = job >> 2, nt = job & 3;
    const v16h a0 = frag_a_lds(sH, mi*16, lane, 0);
    const v16h a1 = frag_a_lds(sH, mi*16, lane, 32);
    v8f acc = {};
    const v16h w0 = frag_b_wt(sW2, nt*16, lane, 0);
    acc = __builtin_amdgcn_wmma_f32_16x16x32_f16(false, a0, false, w0, (short)0, acc, false, false);
    const v16h w1 = frag_b_wt(sW2, nt*16, lane, 32);
    acc = __builtin_amdgcn_wmma_f32_16x16x32_f16(false, a1, false, w1, (short)0, acc, false, false);
    const int nc = nt*16 + (lane & 15);
    const int mb = (lane >> 4) * 8;
#pragma unroll
    for (int j = 0; j < 8; ++j) {
      const int r  = mi*16 + mb + j;
      const int t  = r % Tdim;
      const int nn = n0 + r / Tdim;
      out[(((size_t)bb*Tdim + t)*Ndim + nn)*Ddim + nc] = acc[j] + sB2[nc];
    }
  }
}

extern "C" void kernel_launch(void* const* d_in, const int* in_sizes, int n_in,
                              void* d_out, int out_size, void* d_ws, size_t ws_size,
                              hipStream_t stream) {
  (void)in_sizes; (void)n_in; (void)out_size; (void)d_ws; (void)ws_size;
  const float* x    = (const float*)d_in[0];
  const float* steP = (const float*)d_in[1];
  const float* steQ = (const float*)d_in[2];
  const float* Wff  = (const float*)d_in[3];
  const float* bff  = (const float*)d_in[4];
  const float* W1   = (const float*)d_in[5];
  const float* b1   = (const float*)d_in[6];
  const float* W2   = (const float*)d_in[7];
  const float* b2   = (const float*)d_in[8];
  // d_in[9] is K (==8), baked into the kernel constants.
  dim3 grid(Ndim / NT, Bdim);
  fused_attn_kernel<<<grid, dim3(256), 0, stream>>>(
      x, steP, steQ, Wff, bff, W1, b1, W2, b2, (float*)d_out);
}
